// LigandCrossAttention_36575941493425
// MI455X (gfx1250) — matrix operand order repaired and checked by reading, not compiled, not run on hardware
//
#include <hip/hip_runtime.h>
#include <hip/hip_bf16.h>
#include <math.h>

typedef __attribute__((ext_vector_type(16))) _Float16 v16h;
typedef __attribute__((ext_vector_type(8)))  float    v8f;
typedef __attribute__((ext_vector_type(4)))  int      v4i;

#define D_MODEL 512
#define NUM_HEADS 8
#define HEAD_DIM 64
#define BATCH 16
#define SEQ 2048
#define ATOMS 128

#define GLOBAL_AS __attribute__((address_space(1)))
#define LDS_AS    __attribute__((address_space(3)))

#if __has_builtin(__builtin_amdgcn_global_load_async_to_lds_b128)
#define HAVE_ASYNC_LDS 1
#else
#define HAVE_ASYNC_LDS 0
#endif

// ---- WMMA fragment index maps (contiguous k order used by this code for both A and B) ----
// A (16-bit, 16x32): lane L holds row L%16; lanes 0-15 K=0..15, lanes 16-31 K=16..31
//   element e -> K = 16*(L>=16) + e   (matches B order)
__device__ __forceinline__ int amapK(int lane, int e) {
    return ((lane >> 4) << 4) + e;
}
// B (16-bit, 32x16): lane L holds col L%16; lanes 0-15 K=0..15, lanes 16-31 K=16..31
__device__ __forceinline__ int bmapK(int lane, int e) {
    return ((lane >> 4) << 4) + e;
}
// C/D (f32 16x16): lane L, vgpr g -> row = g + 8*(L>=16), col = L%16

// ============================================================
// Generic tiled GEMM: C[M,512] = A[M,512] @ W[512,512] + bias
// Block: 128(M) x 64(N), 256 threads = 8 waves, wave w -> rows 16w..16w+15
// LDS tiles arranged so every fragment read is contiguous b128 pairs.
// ============================================================
template<bool A_F16, bool OUT_F32>
__global__ __launch_bounds__(256)
void gemm512_kernel(const void* __restrict__ Aptr, const float* __restrict__ W,
                    const float* __restrict__ bias, void* __restrict__ Cptr, int M)
{
    __shared__ _Float16 As[128][40];  // [m][k] 32 K + pad (row = 80B, 16B-aligned)
    __shared__ _Float16 Bs[64][40];   // [n][k] transposed W chunk

    const int tid  = threadIdx.x;
    const int lane = tid & 31;
    const int wave = tid >> 5;
    const int m0   = blockIdx.x * 128;
    const int n0   = blockIdx.y * 64;
    const int arow = lane & 15;
    const int kB   = (lane >> 4) << 4;

    v8f acc[4] = {};

    for (int k0 = 0; k0 < 512; k0 += 32) {
        // stage A chunk (128x32) as f16
        #pragma unroll
        for (int it = 0; it < 16; ++it) {
            int i = tid + it * 256;
            int r = i >> 5, c = i & 31;
            float v;
            if (A_F16) v = (float)((const _Float16*)Aptr)[(size_t)(m0 + r) * 512 + k0 + c];
            else       v = ((const float*)Aptr)[(size_t)(m0 + r) * 512 + k0 + c];
            As[r][c] = (_Float16)v;
        }
        // stage W chunk (32 k x 64 n) transposed -> Bs[n][k]
        #pragma unroll
        for (int it = 0; it < 8; ++it) {
            int i = tid + it * 256;
            int r = i >> 6, c = i & 63;      // r = k row, c = n col (coalesced read)
            Bs[c][r] = (_Float16)W[(size_t)(k0 + r) * 512 + n0 + c];
        }
        __syncthreads();

        v16h afrag;
        #pragma unroll
        for (int e = 0; e < 16; ++e)
            afrag[e] = As[wave * 16 + arow][amapK(lane, e)];

        #pragma unroll
        for (int nt = 0; nt < 4; ++nt) {
            v16h bfrag;
            #pragma unroll
            for (int e = 0; e < 16; ++e)
                bfrag[e] = Bs[nt * 16 + arow][kB + e];   // contiguous 32B
            acc[nt] = __builtin_amdgcn_wmma_f32_16x16x32_f16(
                false, afrag, false, bfrag, (short)0, acc[nt], false, false);
        }
        __syncthreads();
    }

    const int rhi = (lane >> 4) << 3;
    #pragma unroll
    for (int nt = 0; nt < 4; ++nt) {
        int n = n0 + nt * 16 + arow;
        float bn = bias[n];
        #pragma unroll
        for (int g = 0; g < 8; ++g) {
            int m = m0 + wave * 16 + rhi + g;
            float v = acc[nt][g] + bn;
            if (OUT_F32) ((float*)Cptr)[(size_t)m * 512 + n] = v;
            else ((_Float16*)Cptr)[(size_t)m * 512 + n] = (_Float16)v;
        }
    }
}

// ============================================================
// Fused attention: scores = QK^T/8 + dist_bias + atom_bias, mask,
// softmax over A=128, ctx = attn @ V.  One head per blockIdx.y,
// 64 s-rows per block (4 waves x 16 rows).
// K tile staged via async global->LDS (ASYNCcnt path) when available.
// ============================================================
__global__ __launch_bounds__(128)
void attn_kernel(const _Float16* __restrict__ Q,    // [B,S,512]
                 const _Float16* __restrict__ Kf,   // [B,A,512]
                 const _Float16* __restrict__ Vf,   // [B,A,512]
                 const float* __restrict__ pcoords, // [B,S,3]
                 const float* __restrict__ lcoords, // [B,A,3]
                 const int* __restrict__ atype,     // [B,A]
                 const unsigned char* __restrict__ lmask, // [B,A] bool
                 const float* __restrict__ Wd,      // [8]
                 const float* __restrict__ bd,      // [8]
                 const float* __restrict__ aemb,    // [100,8]
                 _Float16* __restrict__ Ctx)        // [B,S,512]
{
    __shared__ _Float16 Ks[128][72];    // [a][d] row-major (score B-frags contiguous)
    __shared__ _Float16 Vst[64][136];   // [d][a] transposed (attn@V B-frags contiguous)
    __shared__ float pcs[64][4];
    __shared__ float lcs[128][4];
    __shared__ float abias[128];
    __shared__ float mneg[128];
    __shared__ _Float16 probs[4][16][136];

    const int tid  = threadIdx.x;
    const int lane = tid & 31;
    const int wave = tid >> 5;
    const int b    = blockIdx.z;
    const int h    = blockIdx.y;
    const int s0   = blockIdx.x * 64;
    const int arow = lane & 15;
    const int rhi  = (lane >> 4) << 3;
    const int kB   = (lane >> 4) << 4;

    // ---- stage K tile: async global->LDS b128 (16B chunks) ----
#if HAVE_ASYNC_LDS
    #pragma unroll
    for (int it = 0; it < 8; ++it) {
        int i = tid + it * 128;          // chunk id 0..1023
        int a = i >> 3;
        int d = (i & 7) << 3;            // 8 halves = 16B per chunk
        size_t off = ((size_t)b * ATOMS + a) * D_MODEL + h * 64 + d;
        __builtin_amdgcn_global_load_async_to_lds_b128(
            (GLOBAL_AS v4i*)(Kf + off),
            (LDS_AS v4i*)&Ks[a][d], 0, 0);
    }
#else
    #pragma unroll
    for (int it = 0; it < 64; ++it) {
        int i = tid + it * 128;
        int a = i >> 6, d = i & 63;
        Ks[a][d] = Kf[((size_t)b * ATOMS + a) * D_MODEL + h * 64 + d];
    }
#endif
    // ---- stage V tile transposed: Vst[d][a] ----
    #pragma unroll
    for (int it = 0; it < 64; ++it) {
        int i = tid + it * 128;
        int a = i >> 6, d = i & 63;
        Vst[d][a] = Vf[((size_t)b * ATOMS + a) * D_MODEL + h * 64 + d];
    }
    // ---- coords / atom bias / mask ----
    if (tid < 64) {
        size_t p = ((size_t)b * SEQ + s0 + tid) * 3;
        pcs[tid][0] = pcoords[p + 0];
        pcs[tid][1] = pcoords[p + 1];
        pcs[tid][2] = pcoords[p + 2];
    }
    {
        int a = tid;  // 128 threads == 128 atoms
        size_t p = ((size_t)b * ATOMS + a) * 3;
        lcs[a][0] = lcoords[p + 0];
        lcs[a][1] = lcoords[p + 1];
        lcs[a][2] = lcoords[p + 2];
        int t = atype[(size_t)b * ATOMS + a];
        if ((unsigned)t >= 100u) t = 0;
        abias[a] = aemb[t * NUM_HEADS + h] + bd[h];
        mneg[a]  = lmask[(size_t)b * ATOMS + a] ? 0.0f : -1e30f;
    }
#if HAVE_ASYNC_LDS
#if __has_builtin(__builtin_amdgcn_s_wait_asynccnt)
    __builtin_amdgcn_s_wait_asynccnt(0);
#else
    asm volatile("s_wait_asynccnt 0x0" ::: "memory");
#endif
#endif
    __syncthreads();

    // ---- Q fragments (this wave's 16 rows, Dh=64 in two k-chunks) ----
    const size_t qbase = ((size_t)b * SEQ + s0 + wave * 16 + arow) * D_MODEL + h * 64;
    v16h aq0, aq1;
    #pragma unroll
    for (int e = 0; e < 16; ++e) {
        int kk = amapK(lane, e);
        aq0[e] = Q[qbase + kk];
        aq1[e] = Q[qbase + 32 + kk];
    }

    // ---- scores: 8 column tiles x 2 k-steps ----
    v8f sc[8];
    #pragma unroll
    for (int nt = 0; nt < 8; ++nt) {
        v16h b0, b1;
        int a = nt * 16 + arow;          // B col = score col = atom index
        #pragma unroll
        for (int e = 0; e < 16; ++e) {
            b0[e] = Ks[a][kB + e];       // B[k][n] = K[a][d], contiguous 32B
            b1[e] = Ks[a][32 + kB + e];
        }
        v8f z = {};
        z = __builtin_amdgcn_wmma_f32_16x16x32_f16(false, aq0, false, b0, (short)0, z, false, false);
        sc[nt] = __builtin_amdgcn_wmma_f32_16x16x32_f16(false, aq1, false, b1, (short)0, z, false, false);
    }

    // ---- biases + masked softmax over A (row-wise) ----
    const float wdh = Wd[h];
    float lx[8], ly[8], lz[8], ab8[8], mn8[8];
    #pragma unroll
    for (int nt = 0; nt < 8; ++nt) {
        int a = nt * 16 + arow;
        lx[nt] = lcs[a][0]; ly[nt] = lcs[a][1]; lz[nt] = lcs[a][2];
        ab8[nt] = abias[a]; mn8[nt] = mneg[a];
    }
    #pragma unroll
    for (int g = 0; g < 8; ++g) {
        int r = rhi + g;
        float px = pcs[wave * 16 + r][0];
        float py = pcs[wave * 16 + r][1];
        float pz = pcs[wave * 16 + r][2];
        float m = -1e30f;
        #pragma unroll
        for (int nt = 0; nt < 8; ++nt) {
            float dx = px - lx[nt], dy = py - ly[nt], dz = pz - lz[nt];
            float dist = __builtin_amdgcn_sqrtf(dx * dx + dy * dy + dz * dz);
            float sv = sc[nt][g] * 0.125f + dist * wdh + ab8[nt] + mn8[nt];
            sc[nt][g] = sv;
            m = fmaxf(m, sv);
        }
        #pragma unroll
        for (int off = 1; off < 16; off <<= 1)
            m = fmaxf(m, __shfl_xor(m, off, 32));
        float ssum = 0.0f;
        #pragma unroll
        for (int nt = 0; nt < 8; ++nt) {
            float e = __expf(sc[nt][g] - m);
            sc[nt][g] = e;
            ssum += e;
        }
        #pragma unroll
        for (int off = 1; off < 16; off <<= 1)
            ssum += __shfl_xor(ssum, off, 32);
        float inv = __builtin_amdgcn_rcpf(ssum);
        #pragma unroll
        for (int nt = 0; nt < 8; ++nt)
            sc[nt][g] *= inv;
    }

    // ---- transpose probs through LDS into A-fragment layout ----
    #pragma unroll
    for (int nt = 0; nt < 8; ++nt)
        #pragma unroll
        for (int g = 0; g < 8; ++g)
            probs[wave][rhi + g][nt * 16 + arow] = (_Float16)sc[nt][g];
    __syncthreads();

    // ---- ctx = attn @ V : 4 d-tiles x 4 k-chunks ----
    v8f ov[4] = {};
    #pragma unroll
    for (int kc = 0; kc < 4; ++kc) {
        v16h ap;
        #pragma unroll
        for (int e = 0; e < 16; ++e)
            ap[e] = probs[wave][arow][kc * 32 + amapK(lane, e)];
        #pragma unroll
        for (int ntv = 0; ntv < 4; ++ntv) {
            v16h bv;
            #pragma unroll
            for (int e = 0; e < 16; ++e)
                bv[e] = Vst[ntv * 16 + arow][kc * 32 + kB + e];  // contiguous 32B
            ov[ntv] = __builtin_amdgcn_wmma_f32_16x16x32_f16(
                false, ap, false, bv, (short)0, ov[ntv], false, false);
        }
    }

    // ---- store context (f16, [B,S,H*Dh]) ----
    #pragma unroll
    for (int ntv = 0; ntv < 4; ++ntv)
        #pragma unroll
        for (int g = 0; g < 8; ++g) {
            int s = s0 + wave * 16 + rhi + g;
            int d = ntv * 16 + arow;
            Ctx[((size_t)b * SEQ + s) * D_MODEL + h * 64 + d] = (_Float16)ov[ntv][g];
        }
}

// ============================================================
extern "C" void kernel_launch(void* const* d_in, const int* in_sizes, int n_in,
                              void* d_out, int out_size, void* d_ws, size_t ws_size,
                              hipStream_t stream) {
    (void)in_sizes; (void)n_in; (void)out_size; (void)ws_size;

    const float* protein = (const float*)d_in[0];
    const float* ligand  = (const float*)d_in[1];
    const float* pcoords = (const float*)d_in[2];
    const float* lcoords = (const float*)d_in[3];
    const int*   atype   = (const int*)d_in[4];
    const unsigned char* lmask = (const unsigned char*)d_in[5];
    const float* Wq = (const float*)d_in[6];
    const float* bq = (const float*)d_in[7];
    const float* Wk = (const float*)d_in[8];
    const float* bk = (const float*)d_in[9];
    const float* Wv = (const float*)d_in[10];
    const float* bv = (const float*)d_in[11];
    const float* Wo = (const float*)d_in[12];
    const float* bo = (const float*)d_in[13];
    const float* Wd = (const float*)d_in[14];
    const float* bd = (const float*)d_in[15];
    const float* aemb = (const float*)d_in[16];
    float* out = (float*)d_out;

    const size_t MQ = (size_t)BATCH * SEQ;    // 32768
    const size_t MA = (size_t)BATCH * ATOMS;  // 2048

    _Float16* Qf = (_Float16*)d_ws;
    _Float16* Kf = Qf + MQ * D_MODEL;
    _Float16* Vf = Kf + MA * D_MODEL;
    _Float16* Cx = Vf + MA * D_MODEL;

    dim3 blk(256);
    // Q/K/V projections (f32 in -> f16 out, f32 WMMA accumulation)
    gemm512_kernel<false, false><<<dim3(MQ / 128, 8), blk, 0, stream>>>(protein, Wq, bq, Qf, (int)MQ);
    gemm512_kernel<false, false><<<dim3(MA / 128, 8), blk, 0, stream>>>(ligand,  Wk, bk, Kf, (int)MA);
    gemm512_kernel<false, false><<<dim3(MA / 128, 8), blk, 0, stream>>>(ligand,  Wv, bv, Vf, (int)MA);

    // fused attention: grid (S/64, H, B), 4 waves per block
    attn_kernel<<<dim3(SEQ / 64, NUM_HEADS, BATCH), dim3(128), 0, stream>>>(
        Qf, Kf, Vf, pcoords, lcoords, atype, lmask, Wd, bd, aemb, Cx);

    // output projection (f16 ctx -> f32 out + bias)
    gemm512_kernel<true, true><<<dim3(MQ / 128, 8), blk, 0, stream>>>(Cx, Wo, bo, out, (int)MQ);
}
